// NeuroTransNet_8091718386014
// MI455X (gfx1250) — matrix operand order repaired and checked by reading, not compiled, not run on hardware
//
#include <hip/hip_runtime.h>
#include <math.h>

typedef _Float16 half_t;
typedef __attribute__((ext_vector_type(16))) _Float16 v16h;
typedef __attribute__((ext_vector_type(8)))  float    v8f;

#define T_LEN 1865
#define B_SZ  64
#define C_IN  22
#define D_MOD 32
#define N_WIN 122
#define K_WIN 50
#define S_WIN 15
#define ROWS  (2*B_SZ*N_WIN)   // 15616 (a-stream rows then v-stream rows)
#define EPSF  1e-5f

// ---------------- workspace layout (float units) ----------------
static constexpr size_t OFF_H2   = 0;                                  // B*32*T
static constexpr size_t SZ_H2    = (size_t)B_SZ*D_MOD*T_LEN;           // 3,819,520
static constexpr size_t OFF_U    = OFF_H2 + SZ_H2;                     // ROWS*32
static constexpr size_t SZ_U     = (size_t)ROWS*32;
static constexpr size_t OFF_R    = OFF_U + SZ_U;
static constexpr size_t OFF_QKV  = OFF_R + SZ_U;                       // ROWS*96
static constexpr size_t SZ_QKV   = (size_t)ROWS*96;
static constexpr size_t OFF_O    = OFF_QKV + SZ_QKV;                   // ROWS*32
static constexpr size_t OFF_FFH  = OFF_O + SZ_U;                       // ROWS*128
static constexpr size_t SZ_FFH   = (size_t)ROWS*128;
static constexpr size_t OFF_FEAT = OFF_FFH + SZ_FFH;                   // B*64*32
static constexpr size_t SZ_FEAT  = (size_t)B_SZ*64*32;
static constexpr size_t OFF_WEFF = OFF_FEAT + SZ_FEAT;                 // 32*22*65
static constexpr size_t SZ_WEFF  = (size_t)32*22*65;                   // 45,760
static constexpr size_t OFF_BEFF = OFF_WEFF + SZ_WEFF;                 // 32
static constexpr size_t OFF_BQKV = OFF_BEFF + 32;                      // 4*96
static constexpr size_t OFF_HALF = OFF_BQKV + 384;                     // f16 weights: 4*12288 halfs

// ================= kernel 0: fold BN1 + branch convs into W_eff, transpose weights to f16 =================
__global__ void prep_kernel(
    const float* ws_w, const float* ws_b, const float* wm_w, const float* wm_b,
    const float* wl_w, const float* wl_b, const float* wx_w, const float* wx_b,
    const float* bn1_g, const float* bn1_b, const float* bn1_m, const float* bn1_v,
    const float* sp_w, const float* sp_b,
    const float* t_wq, const float* t_wk, const float* t_wv, const float* t_wo,
    const float* t_w1, const float* t_w2,
    const float* t_bq, const float* t_bk, const float* t_bv,
    float* weff, float* beff, float* bqkv, half_t* hw)
{
  int gid = blockIdx.x*256 + threadIdx.x;
  if (gid < 45760) {                             // W_eff[oc][c][tau64]
    int tau = gid % 65;
    int c   = (gid/65) % 22;
    int oc  = gid / (65*22);
    float acc = 0.f;
    for (int d = 0; d < 32; ++d) {
      float s = bn1_g[d]*rsqrtf(bn1_v[d]+EPSF);
      const float* wptr; int width, start, dl;
      if      (d <  8) { wptr = ws_w; width = 15; start = 25; dl = d;     }
      else if (d < 16) { wptr = wm_w; width = 25; start = 20; dl = d-8;  }
      else if (d < 24) { wptr = wl_w; width = 51; start = 7;  dl = d-16; }
      else             { wptr = wx_w; width = 65; start = 0;  dl = d-24; }
      int k = tau - start;
      if (k >= 0 && k < width)
        acc += sp_w[(oc*32 + d)*22 + c] * s * wptr[dl*width + k];
    }
    weff[gid] = acc;
  } else if (gid < 45792) {                      // B_eff[oc]
    int oc = gid - 45760;
    float acc = sp_b[oc];
    for (int d = 0; d < 32; ++d) {
      float s = bn1_g[d]*rsqrtf(bn1_v[d]+EPSF);
      float bias_d = (d < 8) ? ws_b[d] : (d < 16) ? wm_b[d-8] : (d < 24) ? wl_b[d-16] : wx_b[d-24];
      float off = bn1_b[d] + s*(bias_d - bn1_m[d]);
      float sc = 0.f;
      for (int c = 0; c < 22; ++c) sc += sp_w[(oc*32 + d)*22 + c];
      acc += off * sc;
    }
    beff[oc] = acc;
  } else if (gid < 45792 + 49152) {              // f16 transposed weights (Nout x K)
    int idx2  = gid - 45792;
    int depth = idx2 / 12288;
    int r     = idx2 % 12288;
    float v;
    if (r < 3072) {                              // wqkv^T: 96 x 32
      int row = r / 32, kk = r % 32;
      if      (row < 32) v = t_wq[(depth*32 + kk)*32 + row];
      else if (row < 64) v = t_wk[(depth*32 + kk)*32 + (row-32)];
      else               v = t_wv[(depth*32 + kk)*32 + (row-64)];
    } else if (r < 4096) {                       // wo^T: 32 x 32
      int rr = r - 3072, row = rr/32, kk = rr%32;
      v = t_wo[(depth*32 + kk)*32 + row];
    } else if (r < 8192) {                       // w1^T: 128 x 32
      int rr = r - 4096, row = rr/32, kk = rr%32;
      v = t_w1[(depth*32 + kk)*128 + row];
    } else {                                     // w2^T: 32 x 128
      int rr = r - 8192, row = rr/128, kk = rr%128;
      v = t_w2[(depth*128 + kk)*32 + row];
    }
    hw[(size_t)depth*12288 + r] = (half_t)v;
  } else if (gid < 45792 + 49152 + 384) {        // fused qkv bias
    int j2 = gid - (45792 + 49152);
    int depth = j2 / 96, j = j2 % 96;
    float v = (j < 32) ? t_bq[depth*32 + j] : (j < 64) ? t_bk[depth*32 + j-32] : t_bv[depth*32 + j-64];
    bqkv[j2] = v;
  }
}

// ================= kernel 1: fused effective conv + BN2 + ELU -> h2 (B,32,T) =================
__global__ void conv_front_kernel(const float* __restrict__ x, const float* __restrict__ weff,
                                  const float* __restrict__ beff,
                                  const float* bn2_g, const float* bn2_b,
                                  const float* bn2_m, const float* bn2_v,
                                  float* __restrict__ h2)
{
  __shared__ float xs[C_IN*128];                 // x[b, c, t0-32 .. t0+95]
  int b  = blockIdx.y;
  int t0 = blockIdx.x * 64;
  for (int idx = threadIdx.x; idx < C_IN*128; idx += 256) {
    int c = idx >> 7, i = idx & 127;
    int t = t0 - 32 + i;
    xs[idx] = (t >= 0 && t < T_LEN) ? x[(size_t)(b*C_IN + c)*T_LEN + t] : 0.f;
  }
  __syncthreads();
  int oc = threadIdx.x & 31;
  int tb = threadIdx.x >> 5;                     // 0..7
  float acc[8];
  float be = beff[oc];
  #pragma unroll
  for (int i = 0; i < 8; ++i) acc[i] = be;
  const float* wrow = weff + (size_t)oc*C_IN*65;
  for (int c = 0; c < C_IN; ++c) {
    const float* xc = xs + c*128 + tb;
    const float* wc = wrow + c*65;
    for (int tau = 0; tau < 65; ++tau) {
      float w = wc[tau];
      #pragma unroll
      for (int i = 0; i < 8; ++i) acc[i] += w * xc[tau + 8*i];
    }
  }
  float s2 = bn2_g[oc]*rsqrtf(bn2_v[oc]+EPSF);
  float o2 = bn2_b[oc] - s2*bn2_m[oc];
  #pragma unroll
  for (int i = 0; i < 8; ++i) {
    int t = t0 + tb + 8*i;
    if (t < T_LEN) {
      float v = s2*acc[i] + o2;
      h2[(size_t)(b*D_MOD + oc)*T_LEN + t] = (v > 0.f) ? v : expm1f(v);
    }
  }
}

// ================= kernel 2: window mean / log-var -> u (a rows then v rows) =================
__global__ void window_kernel(const float* __restrict__ h2, float* __restrict__ u)
{
  int idx = blockIdx.x*256 + threadIdx.x;
  if (idx >= B_SZ*N_WIN*D_MOD) return;
  int oc = idx & 31;
  int n  = (idx >> 5) % N_WIN;
  int b  = idx / (32*N_WIN);
  const float* p = h2 + (size_t)(b*D_MOD + oc)*T_LEN + n*S_WIN;
  float s = 0.f, ss = 0.f;
  for (int k = 0; k < K_WIN; ++k) { float v = p[k]; s += v; ss += v*v; }
  float mu  = s * (1.f/K_WIN);
  float var = (ss - K_WIN*mu*mu) * (1.f/(K_WIN-1));
  var = fminf(fmaxf(var, 1e-6f), 1e6f);
  int row = b*N_WIN + n;
  u[(size_t)row*D_MOD + oc]              = mu;
  u[(size_t)(ROWS/2 + row)*D_MOD + oc]   = logf(var);
}

// ================= kernel 3: LayerNorm (D=32, one wave32 per row) =================
__global__ void ln_kernel(const float* __restrict__ z, const float* g, const float* bb,
                          float* __restrict__ r, int M)
{
  int lane = threadIdx.x & 31;
  int row  = blockIdx.x*4 + (threadIdx.x >> 5);
  if (row >= M) return;
  float v = z[(size_t)row*32 + lane];
  float s = v;
  #pragma unroll
  for (int o = 16; o >= 1; o >>= 1) s += __shfl_xor(s, o, 32);
  float mu = s * (1.f/32.f);
  float d  = v - mu;
  float q  = d*d;
  #pragma unroll
  for (int o = 16; o >= 1; o >>= 1) q += __shfl_xor(q, o, 32);
  float va = q * (1.f/32.f);
  r[(size_t)row*32 + lane] = d * rsqrtf(va + EPSF) * g[lane] + bb[lane];
}

// ================= WMMA fragment loaders (CDNA5 16-bit layouts, wave32) =================
// A 16x32: lanes 0-15: VGPR0-3 => K 0..7, VGPR4-7 => K 16..23; lanes 16-31: K 8..15 / 24..31
__device__ __forceinline__ v16h load_frag_a(const half_t* base, int K, int lane) {
  int row = lane & 15, hf = lane >> 4;
  v16h f;
  #pragma unroll
  for (int j = 0; j < 8; ++j) {
    int k0 = (j < 4) ? (hf*8 + 2*j) : (16 + hf*8 + 2*(j-4));
    f[2*j]   = base[row*K + k0];
    f[2*j+1] = base[row*K + k0 + 1];
  }
  return f;
}
// B 32x16: lanes 0-15: VGPR0-3 => K 0..7, VGPR4-7 => K 16..23; lanes 16-31: K 8..15 / 24..31
__device__ __forceinline__ v16h load_frag_b(const half_t* base, int K, int lane) {
  int col = lane & 15, hf = lane >> 4;
  v16h f;
  #pragma unroll
  for (int j = 0; j < 8; ++j) {
    int k0 = (j < 4) ? (hf*8 + 2*j) : (16 + hf*8 + 2*(j-4));
    f[2*j]   = base[col*K + k0];
    f[2*j+1] = base[col*K + k0 + 1];
  }
  return f;
}

// ================= kernel 4: WMMA GEMM  C = A(MxK) * Wt^T(NoutxK) + bias [+resid] [gelu] ======
__global__ void gemm_wmma_kernel(const float* __restrict__ A, const half_t* __restrict__ Wt,
                                 const float* __restrict__ bias, const float* __restrict__ resid,
                                 float* __restrict__ C, int M, int K, int Nout, int act)
{
  __shared__ half_t sm[4*2*16*128];              // 4 waves x (A tile + B tile), K<=128
  int lane = threadIdx.x & 31;
  int wv   = threadIdx.x >> 5;
  int ntiles = Nout >> 4;
  int total  = (M >> 4) * ntiles;
  int tile   = blockIdx.x*4 + wv;
  if (tile >= total) return;
  int nt = tile % ntiles, mt = tile / ntiles;
  int m0 = mt << 4, n0 = nt << 4;
  half_t* As = sm + wv*(2*16*128);
  half_t* Bs = As + 16*128;
  for (int idx = lane; idx < 16*K; idx += 32) {   // stage + f16-convert (intra-wave, no barrier)
    int rr = idx / K, cc = idx - rr*K;
    As[rr*K + cc] = (half_t)A[(size_t)(m0+rr)*K + cc];
    Bs[rr*K + cc] = Wt[(size_t)(n0+rr)*K + cc];
  }
  v8f acc = {};
  for (int kb = 0; kb < K; kb += 32) {
    v16h af = load_frag_a(As + kb, K, lane);
    v16h bf = load_frag_b(Bs + kb, K, lane);
    acc = __builtin_amdgcn_wmma_f32_16x16x32_f16(false, af, false, bf, (short)0, acc, false, false);
  }
  int n = lane & 15, hf = lane >> 4;
  float bv = bias[n0 + n];
  #pragma unroll
  for (int rv = 0; rv < 8; ++rv) {
    int m = m0 + hf*8 + rv;                       // C/D layout: VGPR rv -> row rv + 8*half
    size_t o = (size_t)m*Nout + n0 + n;
    float v = acc[rv] + bv;
    if (resid) v += resid[o];
    if (act)   v = 0.5f*v*(1.f + erff(v*0.70710678118f));   // exact GELU
    C[o] = v;
  }
}

// ================= kernel 5: attention (dk=4, N=122), one block per (stream,b,head) ==========
__global__ void attn_kernel(const float* __restrict__ qkv, float* __restrict__ obuf)
{
  __shared__ float ks[N_WIN*4], vs[N_WIN*4];
  int flat = blockIdx.x;                          // flat = s*512 + b*8 + h
  int h = flat & 7, b = (flat >> 3) & 63, s = flat >> 9;
  int rowbase = s*(B_SZ*N_WIN) + b*N_WIN;
  int colq = h*4;
  for (int idx = threadIdx.x; idx < N_WIN*4; idx += 128) {
    int n = idx >> 2, dd = idx & 3;
    ks[idx] = qkv[(size_t)(rowbase+n)*96 + 32 + colq + dd];
    vs[idx] = qkv[(size_t)(rowbase+n)*96 + 64 + colq + dd];
  }
  __syncthreads();
  int qi = threadIdx.x;
  if (qi >= N_WIN) return;
  const float* qp = qkv + (size_t)(rowbase+qi)*96 + colq;
  float q0 = qp[0], q1 = qp[1], q2 = qp[2], q3 = qp[3];
  float mx = -1e30f;
  for (int kk = 0; kk < N_WIN; ++kk) {
    float sc = 0.5f*(q0*ks[kk*4] + q1*ks[kk*4+1] + q2*ks[kk*4+2] + q3*ks[kk*4+3]);
    mx = fmaxf(mx, sc);
  }
  float sum = 0.f, o0 = 0.f, o1 = 0.f, o2 = 0.f, o3 = 0.f;
  for (int kk = 0; kk < N_WIN; ++kk) {
    float sc = 0.5f*(q0*ks[kk*4] + q1*ks[kk*4+1] + q2*ks[kk*4+2] + q3*ks[kk*4+3]);
    float e = expf(sc - mx);
    sum += e;
    o0 += e*vs[kk*4]; o1 += e*vs[kk*4+1]; o2 += e*vs[kk*4+2]; o3 += e*vs[kk*4+3];
  }
  float inv = 1.f/sum;
  float* op = obuf + (size_t)(rowbase+qi)*32 + colq;
  op[0] = o0*inv; op[1] = o1*inv; op[2] = o2*inv; op[3] = o3*inv;
}

// ================= kernel 6: feature conv + BN3 + ELU -> feat (B,64,32) ======================
__global__ void feconv_kernel(const float* __restrict__ u, const float* __restrict__ fe_w,
                              const float* fe_b, const float* g, const float* bb,
                              const float* m, const float* vv, float* __restrict__ feat)
{
  int idx = blockIdx.x*256 + threadIdx.x;
  if (idx >= B_SZ*64*32) return;
  int dd = idx & 31, oc = (idx >> 5) & 63, b = idx >> 11;
  float acc = fe_b[oc];
  for (int n = 0; n < N_WIN; ++n) {
    float wa = fe_w[(oc*N_WIN + n)*2 + 0];
    float wv = fe_w[(oc*N_WIN + n)*2 + 1];
    acc += wa * u[(size_t)(b*N_WIN + n)*32 + dd]
         + wv * u[(size_t)(ROWS/2 + b*N_WIN + n)*32 + dd];
  }
  float s3 = g[oc]*rsqrtf(vv[oc]+EPSF);
  float v  = s3*(acc - m[oc]) + bb[oc];
  feat[idx] = (v > 0.f) ? v : expm1f(v);
}

// ================= kernel 7: classifier (B,2048) @ (2048,4) ================================
__global__ void cls_kernel(const float* __restrict__ feat, const float* __restrict__ cls_w,
                           const float* cls_b, float* __restrict__ out)
{
  int tid = threadIdx.x;
  if (tid >= 256) return;
  int b = tid >> 2, nc = tid & 3;
  float acc = cls_b[nc];
  const float* f = feat + (size_t)b*2048;
  for (int j = 0; j < 2048; ++j) acc += f[j] * cls_w[j*4 + nc];
  out[b*4 + nc] = acc;
}

// =============================================================================================
extern "C" void kernel_launch(void* const* d_in, const int* in_sizes, int n_in,
                              void* d_out, int out_size, void* d_ws, size_t ws_size,
                              hipStream_t stream)
{
  const float* x     = (const float*)d_in[0];
  const float* ws_w  = (const float*)d_in[1];  const float* ws_b = (const float*)d_in[2];
  const float* wm_w  = (const float*)d_in[3];  const float* wm_b = (const float*)d_in[4];
  const float* wl_w  = (const float*)d_in[5];  const float* wl_b = (const float*)d_in[6];
  const float* wx_w  = (const float*)d_in[7];  const float* wx_b = (const float*)d_in[8];
  const float* bn1_g = (const float*)d_in[9];  const float* bn1_b = (const float*)d_in[10];
  const float* bn1_m = (const float*)d_in[11]; const float* bn1_v = (const float*)d_in[12];
  const float* bn2_g = (const float*)d_in[13]; const float* bn2_b = (const float*)d_in[14];
  const float* bn2_m = (const float*)d_in[15]; const float* bn2_v = (const float*)d_in[16];
  const float* bn3_g = (const float*)d_in[17]; const float* bn3_b = (const float*)d_in[18];
  const float* bn3_m = (const float*)d_in[19]; const float* bn3_v = (const float*)d_in[20];
  const float* sp_w  = (const float*)d_in[21]; const float* sp_b  = (const float*)d_in[22];
  const float* ln1_g = (const float*)d_in[23]; const float* ln1_b = (const float*)d_in[24];
  const float* ln2_g = (const float*)d_in[25]; const float* ln2_b = (const float*)d_in[26];
  const float* t_wq  = (const float*)d_in[27]; const float* t_bq  = (const float*)d_in[28];
  const float* t_wk  = (const float*)d_in[29]; const float* t_bk  = (const float*)d_in[30];
  const float* t_wv  = (const float*)d_in[31]; const float* t_bv  = (const float*)d_in[32];
  const float* t_wo  = (const float*)d_in[33]; const float* t_bo  = (const float*)d_in[34];
  const float* t_w1  = (const float*)d_in[35]; const float* t_b1  = (const float*)d_in[36];
  const float* t_w2  = (const float*)d_in[37]; const float* t_b2  = (const float*)d_in[38];
  const float* fe_w  = (const float*)d_in[39]; const float* fe_b  = (const float*)d_in[40];
  const float* cls_w = (const float*)d_in[41]; const float* cls_b = (const float*)d_in[42];

  float*  ws   = (float*)d_ws;
  float*  h2   = ws + OFF_H2;
  float*  u    = ws + OFF_U;
  float*  r    = ws + OFF_R;
  float*  qkv  = ws + OFF_QKV;
  float*  ob   = ws + OFF_O;
  float*  ffh  = ws + OFF_FFH;
  float*  feat = ws + OFF_FEAT;
  float*  weff = ws + OFF_WEFF;
  float*  beff = ws + OFF_BEFF;
  float*  bqkv = ws + OFF_BQKV;
  half_t* hw   = (half_t*)(ws + OFF_HALF);
  float*  out  = (float*)d_out;

  // 0) fold + transpose weights
  prep_kernel<<<(45792 + 49152 + 384 + 255)/256, 256, 0, stream>>>(
      ws_w, ws_b, wm_w, wm_b, wl_w, wl_b, wx_w, wx_b,
      bn1_g, bn1_b, bn1_m, bn1_v, sp_w, sp_b,
      t_wq, t_wk, t_wv, t_wo, t_w1, t_w2, t_bq, t_bk, t_bv,
      weff, beff, bqkv, hw);

  // 1) fused conv front-end
  conv_front_kernel<<<dim3((T_LEN + 63)/64, B_SZ), 256, 0, stream>>>(
      x, weff, beff, bn2_g, bn2_b, bn2_m, bn2_v, h2);

  // 2) window stats -> token matrix u (ROWS x 32)
  window_kernel<<<(B_SZ*N_WIN*D_MOD + 255)/256, 256, 0, stream>>>(h2, u);

  // 3) transformer: a and v streams batched (M = 15616)
  const int M = ROWS;
  for (int i = 0; i < 4; ++i) {
    const half_t* wqkvt = hw + (size_t)i*12288;
    const half_t* wot   = wqkvt + 3072;
    const half_t* w1t   = wqkvt + 4096;
    const half_t* w2t   = wqkvt + 8192;

    ln_kernel<<<M/4, 128, 0, stream>>>(u, ln1_g + i*32, ln1_b + i*32, r, M);
    gemm_wmma_kernel<<<((M/16)*6 + 3)/4, 128, 0, stream>>>(
        r, wqkvt, bqkv + i*96, nullptr, qkv, M, 32, 96, 0);
    attn_kernel<<<2*B_SZ*8, 128, 0, stream>>>(qkv, ob);
    gemm_wmma_kernel<<<((M/16)*2 + 3)/4, 128, 0, stream>>>(
        ob, wot, t_bo + i*32, u, u, M, 32, 32, 0);          // residual, in-place
    ln_kernel<<<M/4, 128, 0, stream>>>(u, ln2_g + i*32, ln2_b + i*32, r, M);
    gemm_wmma_kernel<<<((M/16)*8 + 3)/4, 128, 0, stream>>>(
        r, w1t, t_b1 + i*128, nullptr, ffh, M, 32, 128, 1); // fused GELU
    gemm_wmma_kernel<<<((M/16)*2 + 3)/4, 128, 0, stream>>>(
        ffh, w2t, t_b2 + i*32, u, u, M, 128, 32, 0);        // residual, in-place
  }

  // 4) feature conv + BN3 + ELU
  feconv_kernel<<<(B_SZ*64*32 + 255)/256, 256, 0, stream>>>(
      u, fe_w, fe_b, bn3_g, bn3_b, bn3_m, bn3_v, feat);

  // 5) classifier
  cls_kernel<<<1, 256, 0, stream>>>(feat, cls_w, cls_b, out);
}
